// MPNNClassifier_46557445489257
// MI455X (gfx1250) — matrix operand order repaired and checked by reading, not compiled, not run on hardware
//
#include <hip/hip_runtime.h>
#include <hip/hip_bf16.h>
#include <stdint.h>

// ---------------- CDNA5 WMMA types ----------------
typedef __bf16 bf16_t;
typedef __attribute__((ext_vector_type(16))) __bf16 v16bf;
typedef __attribute__((ext_vector_type(8)))  float  v8f;

#define N_NODES 100000
#define N_EDGES 250000
#define N_GRAPHS 512
#define EMBED 256
#define HIDDEN 512
#define HID2 256
#define OUTF 128

// ---------------------------------------------------------------------------
// CDNA5 async global->LDS copy (ASYNCcnt) helpers
// ---------------------------------------------------------------------------
__device__ __forceinline__ void async_load_b128(uint32_t lds_byte_off, const void* gaddr) {
  asm volatile("global_load_async_to_lds_b128 %0, %1, off"
               :: "v"(lds_byte_off), "v"(gaddr) : "memory");
}
__device__ __forceinline__ void wait_asynccnt0() {
  asm volatile("s_wait_asynccnt 0x0" ::: "memory");
}

// ---------------------------------------------------------------------------
// Generic zero / copy (float4 where possible)
// ---------------------------------------------------------------------------
__global__ void k_zero(float* __restrict__ p, size_t n) {
  size_t i = blockIdx.x * (size_t)blockDim.x + threadIdx.x;
  if (i < n) p[i] = 0.0f;
}

__global__ void k_copy4(const float4* __restrict__ s, float4* __restrict__ d, size_t n4) {
  size_t i = blockIdx.x * (size_t)blockDim.x + threadIdx.x;
  if (i < n4) d[i] = s[i];
}

// ---------------------------------------------------------------------------
// Embedding gather: H[n, :] = emb[x[n], :]  (float4 over features)
// ---------------------------------------------------------------------------
__global__ void k_embed(const int* __restrict__ x, const float* __restrict__ emb,
                        float* __restrict__ H, int M, int F) {
  size_t i = blockIdx.x * (size_t)blockDim.x + threadIdx.x;
  int F4 = F >> 2;
  if (i >= (size_t)M * F4) return;
  int n  = (int)(i / F4);
  int f4 = (int)(i % F4) << 2;
  *(float4*)&H[(size_t)n * F + f4] = *(const float4*)&emb[(size_t)x[n] * F + f4];
}

// ---------------------------------------------------------------------------
// Edge aggregation: AGG[dst[e], f..f+3] += H[src[e], f..f+3]
// ---------------------------------------------------------------------------
__global__ void k_edge_agg(const float* __restrict__ H, const int* __restrict__ src,
                           const int* __restrict__ dst, float* __restrict__ AGG,
                           int E, int F) {
  size_t i = blockIdx.x * (size_t)blockDim.x + threadIdx.x;
  int F4 = F >> 2;
  if (i >= (size_t)E * F4) return;
  int e  = (int)(i / F4);
  int f4 = (int)(i % F4) << 2;
  float4 v = *(const float4*)&H[(size_t)src[e] * F + f4];
  float* p = &AGG[(size_t)dst[e] * F + f4];
  atomicAdd(p + 0, v.x);
  atomicAdd(p + 1, v.y);
  atomicAdd(p + 2, v.z);
  atomicAdd(p + 3, v.w);
}

// ---------------------------------------------------------------------------
// Weight prep: Wt[n, k] = bf16(W[k, n])   (transpose + convert, done once)
// ---------------------------------------------------------------------------
__global__ void k_prep_wt(const float* __restrict__ W, bf16_t* __restrict__ Wt,
                          int K, int N) {
  size_t i = blockIdx.x * (size_t)blockDim.x + threadIdx.x;
  if (i >= (size_t)K * N) return;
  int k = (int)(i / N);
  int n = (int)(i % N);
  Wt[(size_t)n * K + k] = (bf16_t)W[i];
}

// ---------------------------------------------------------------------------
// bf16 WMMA GEMM: C[M,N] = A[M,K] @ B[K,N] (+bias)(+relu), B given as Bt[N][K] bf16.
//
// Block = 128 threads (4 waves), covers 16 rows x 256 cols.
// Per wave: 4 x (16x16) accumulators sharing one A fragment per K-step.
// A tile (16x32 f32, 2KB) staged to LDS via global_load_async_to_lds_b128.
//
// Fragment layouts used here (wave32; A and B in contiguous K order, not the ISA table order):
//  A 16x32 bf16 : lanes 0-15 row M=l  K 0..15;         lanes 16-31 row M=l-16 K 16..31
//  B 32x16 bf16 : lanes 0-15 col N=l  K 0..15;         lanes 16-31 col N=l-16 K 16..31
//  D 16x16 f32  : lanes 0-15 col N=l rows 0..7;        lanes 16-31 col N=l-16 rows 8..15
// ---------------------------------------------------------------------------
__global__ __launch_bounds__(128)
void k_gemm_wmma(const float* __restrict__ A, const bf16_t* __restrict__ Bt,
                 const float* __restrict__ bias, float* __restrict__ C,
                 int M, int N, int K, int relu) {
  __shared__ float As[16 * 32];

  const int tid  = threadIdx.x;
  const int wave = tid >> 5;
  const int lane = tid & 31;
  const int tm   = blockIdx.y * 16;              // M tile base (always valid)
  const int tn   = blockIdx.x * 256 + wave * 64; // this wave's N strip base

  const int half = lane >> 4;   // 0 or 1
  const int l16  = lane & 15;
  const int ak0  = half * 16;   // A K-base within 32-slice
  const int bk0  = half * 16;   // B K-base within 32-slice

  // async stage parameters: thread -> (row, float4 chunk) of the 16x32 A tile
  const int arow = tid >> 3;          // 0..15
  const int ac4  = (tid & 7) << 2;    // 0,4,...,28
  const uint32_t lds_off = (uint32_t)(uintptr_t)&As[arow * 32 + ac4];

  v8f acc[4] = {v8f{}, v8f{}, v8f{}, v8f{}};

  for (int kt = 0; kt < K; kt += 32) {
    // ---- stage A tile into LDS (128 x b128 async copies = 2KB) ----
    async_load_b128(lds_off, A + (size_t)(tm + arow) * K + kt + ac4);
    wait_asynccnt0();
    __syncthreads();

    // ---- build A fragment from LDS, convert f32 -> bf16 ----
    v16bf a;
    const float* Ap = &As[l16 * 32 + ak0];
#pragma unroll
    for (int j = 0; j < 8; ++j) {
      a[j]     = (bf16_t)Ap[j];        // K = kt+ak0 + 0..7
      a[j + 8] = (bf16_t)Ap[j + 8];    // K = kt+ak0 + 8..15
    }

    // ---- 4 N-tiles: contiguous 32B bf16 B fragments, 4 WMMAs ----
#pragma unroll
    for (int t = 0; t < 4; ++t) {
      int cn = tn + t * 16;
      int cc = (cn < N) ? cn : (N - 16);  // clamp (wave stays converged; store guarded)
      v16bf b = *(const v16bf*)&Bt[(size_t)(cc + l16) * K + kt + bk0];
      acc[t] = __builtin_amdgcn_wmma_f32_16x16x32_bf16(
          false, a, false, b, (short)0, acc[t], false, false);
    }
    __syncthreads();  // protect As before next iteration's async store
  }

  // ---- epilogue: bias (+relu), store ----
  const int crow0 = tm + half * 8;
#pragma unroll
  for (int t = 0; t < 4; ++t) {
    int cn = tn + t * 16;
    if (cn >= N) continue;
    int ccol = cn + l16;
    float bv = bias ? bias[ccol] : 0.0f;
#pragma unroll
    for (int i = 0; i < 8; ++i) {
      float v = acc[t][i] + bv;
      if (relu) v = fmaxf(v, 0.0f);
      C[(size_t)(crow0 + i) * N + ccol] = v;
    }
  }
}

// ---------------------------------------------------------------------------
// BatchNorm stats: per-column sum / sumsq partials + atomics
// ---------------------------------------------------------------------------
__global__ void k_bn_stats(const float* __restrict__ H, float* __restrict__ stats,
                           int M, int N) {
  int col = blockIdx.x * blockDim.x + threadIdx.x;
  if (col >= N) return;
  int rowsPer = (M + gridDim.y - 1) / gridDim.y;
  int r0 = blockIdx.y * rowsPer;
  int r1 = min(M, r0 + rowsPer);
  float s = 0.0f, ss = 0.0f;
  for (int r = r0; r < r1; ++r) {
    float v = H[(size_t)r * N + col];
    s += v; ss += v * v;
  }
  atomicAdd(&stats[col], s);
  atomicAdd(&stats[N + col], ss);
}

__global__ void k_bn_apply(float* __restrict__ H, const float* __restrict__ stats,
                           const float* __restrict__ g, const float* __restrict__ be,
                           int M, int N) {
  size_t i = blockIdx.x * (size_t)blockDim.x + threadIdx.x;
  if (i >= (size_t)M * N) return;
  int col = (int)(i % N);
  float invM = 1.0f / (float)M;
  float mu  = stats[col] * invM;
  float var = stats[N + col] * invM - mu * mu;
  float v = (H[i] - mu) * rsqrtf(var + 1e-5f) * g[col] + be[col];
  H[i] = fmaxf(v, 0.0f);
}

// ---------------------------------------------------------------------------
// Pooling (batch is sorted -> segmented running sums, few atomics)
// ---------------------------------------------------------------------------
__global__ void k_count(const int* __restrict__ batch, float* __restrict__ cnt, int M) {
  int n = blockIdx.x * blockDim.x + threadIdx.x;
  if (n < M) atomicAdd(&cnt[batch[n]], 1.0f);
}

__global__ void k_pool_sum(const float* __restrict__ H, const int* __restrict__ batch,
                           float* __restrict__ pooled, int M, int F) {
  int col = blockIdx.x * blockDim.x + threadIdx.x;
  if (col >= F) return;
  int r0 = blockIdx.y * 128;
  int r1 = min(M, r0 + 128);
  if (r0 >= M) return;
  float acc = 0.0f;
  int cur = batch[r0];                 // wave-uniform load (broadcast)
  for (int r = r0; r < r1; ++r) {
    int b = batch[r];
    if (b != cur) {
      atomicAdd(&pooled[(size_t)cur * F + col], acc);
      acc = 0.0f; cur = b;
    }
    acc += H[(size_t)r * F + col];
  }
  atomicAdd(&pooled[(size_t)cur * F + col], acc);
}

__global__ void k_pool_div(float* __restrict__ pooled, const float* __restrict__ cnt,
                           int G, int F) {
  size_t i = blockIdx.x * (size_t)blockDim.x + threadIdx.x;
  if (i >= (size_t)G * F) return;
  int g = (int)(i / F);
  pooled[i] /= fmaxf(cnt[g], 1.0f);
}

// ---------------------------------------------------------------------------
// Host-side orchestration
// ---------------------------------------------------------------------------
static inline dim3 grid1d(size_t n, int b) { return dim3((unsigned)((n + b - 1) / b)); }

extern "C" void kernel_launch(void* const* d_in, const int* in_sizes, int n_in,
                              void* d_out, int out_size, void* d_ws, size_t ws_size,
                              hipStream_t stream) {
  const int*   x     = (const int*)  d_in[0];
  const int*   ei    = (const int*)  d_in[1];   // (2, E): src = ei, dst = ei+E
  const int*   batch = (const int*)  d_in[2];
  const float* emb   = (const float*)d_in[3];
  const float* W1    = (const float*)d_in[4];
  const float* b1    = (const float*)d_in[5];
  const float* g1    = (const float*)d_in[6];
  const float* be1   = (const float*)d_in[7];
  const float* W2    = (const float*)d_in[8];
  const float* b2    = (const float*)d_in[9];
  const float* g2    = (const float*)d_in[10];
  const float* be2   = (const float*)d_in[11];
  const float* Wl1   = (const float*)d_in[12];
  const float* bl1   = (const float*)d_in[13];
  const float* Wl2   = (const float*)d_in[14];
  const float* bl2   = (const float*)d_in[15];
  float* out = (float*)d_out;

  const int* srcI = ei;
  const int* dstI = ei + N_EDGES;

  // ---- workspace layout (float offsets; all bf16 regions 32B-aligned) ----
  float* ws = (float*)d_ws;
  size_t off = 0;
  float* bufA   = ws + off; off += (size_t)N_NODES * HIDDEN;   // node features
  float* bufB   = ws + off; off += (size_t)N_NODES * HIDDEN;   // aggregates
  float* stats  = ws + off; off += 2 * HIDDEN;                 // [sum|sumsq]
  float* pooled = ws + off; off += (size_t)N_GRAPHS * HIDDEN;  // (512,512)
  float* cnts   = ws + off; off += N_GRAPHS;
  float* z      = ws + off; off += (size_t)N_GRAPHS * HID2;    // (512,256)
  off = (off + 7) & ~(size_t)7;                                // 32B align
  bf16_t* Wt1  = (bf16_t*)(ws + off); off += (size_t)EMBED  * HIDDEN / 2; // [512][256]
  bf16_t* Wt2  = (bf16_t*)(ws + off); off += (size_t)HIDDEN * HIDDEN / 2; // [512][512]
  bf16_t* Wtl1 = (bf16_t*)(ws + off); off += (size_t)HIDDEN * HID2   / 2; // [256][512]
  bf16_t* Wtl2 = (bf16_t*)(ws + off); off += (size_t)HID2   * OUTF   / 2; // [128][256]

  const int B = 256;

  // ---- one-time weight transpose+convert to bf16 [N][K] ----
  k_prep_wt<<<grid1d((size_t)EMBED * HIDDEN, B), B, 0, stream>>>(W1, Wt1, EMBED, HIDDEN);
  k_prep_wt<<<grid1d((size_t)HIDDEN * HIDDEN, B), B, 0, stream>>>(W2, Wt2, HIDDEN, HIDDEN);
  k_prep_wt<<<grid1d((size_t)HIDDEN * HID2, B), B, 0, stream>>>(Wl1, Wtl1, HIDDEN, HID2);
  k_prep_wt<<<grid1d((size_t)HID2 * OUTF, B), B, 0, stream>>>(Wl2, Wtl2, HID2, OUTF);

  // ---- layer 0: embedding gather -> bufA (M x 256) ----
  size_t n0 = (size_t)N_NODES * EMBED;
  k_embed<<<grid1d(n0 / 4, B), B, 0, stream>>>(x, emb, bufA, N_NODES, EMBED);

  // ---- MPNN layer 1: agg = h + segsum(h[src] -> dst) ----
  k_copy4<<<grid1d(n0 / 4, B), B, 0, stream>>>((const float4*)bufA, (float4*)bufB, n0 / 4);
  k_edge_agg<<<grid1d((size_t)N_EDGES * EMBED / 4, B), B, 0, stream>>>(
      bufA, srcI, dstI, bufB, N_EDGES, EMBED);

  // GEMM1: bufB (M x 256) @ W1 -> bufA (M x 512)
  {
    dim3 g(HIDDEN / 256, N_NODES / 16);
    k_gemm_wmma<<<g, 128, 0, stream>>>(bufB, Wt1, b1, bufA, N_NODES, HIDDEN, EMBED, 0);
  }
  // BN1 + ReLU in place
  k_zero<<<grid1d(2 * HIDDEN, B), B, 0, stream>>>(stats, 2 * HIDDEN);
  { dim3 g(HIDDEN / 256, 128); k_bn_stats<<<g, 256, 0, stream>>>(bufA, stats, N_NODES, HIDDEN); }
  size_t n1 = (size_t)N_NODES * HIDDEN;
  k_bn_apply<<<grid1d(n1, B), B, 0, stream>>>(bufA, stats, g1, be1, N_NODES, HIDDEN);

  // ---- MPNN layer 2 ----
  k_copy4<<<grid1d(n1 / 4, B), B, 0, stream>>>((const float4*)bufA, (float4*)bufB, n1 / 4);
  k_edge_agg<<<grid1d((size_t)N_EDGES * HIDDEN / 4, B), B, 0, stream>>>(
      bufA, srcI, dstI, bufB, N_EDGES, HIDDEN);

  // GEMM2: bufB (M x 512) @ W2 -> bufA (M x 512)
  {
    dim3 g(HIDDEN / 256, N_NODES / 16);
    k_gemm_wmma<<<g, 128, 0, stream>>>(bufB, Wt2, b2, bufA, N_NODES, HIDDEN, HIDDEN, 0);
  }
  k_zero<<<grid1d(2 * HIDDEN, B), B, 0, stream>>>(stats, 2 * HIDDEN);
  { dim3 g(HIDDEN / 256, 128); k_bn_stats<<<g, 256, 0, stream>>>(bufA, stats, N_NODES, HIDDEN); }
  k_bn_apply<<<grid1d(n1, B), B, 0, stream>>>(bufA, stats, g2, be2, N_NODES, HIDDEN);

  // ---- mean pool per graph ----
  size_t np = (size_t)N_GRAPHS * HIDDEN + N_GRAPHS;  // pooled + cnts contiguous
  k_zero<<<grid1d(np, B), B, 0, stream>>>(pooled, np);
  k_count<<<grid1d(N_NODES, B), B, 0, stream>>>(batch, cnts, N_NODES);
  {
    dim3 g(HIDDEN / 256, (N_NODES + 127) / 128);
    k_pool_sum<<<g, 256, 0, stream>>>(bufA, batch, pooled, N_NODES, HIDDEN);
  }
  k_pool_div<<<grid1d((size_t)N_GRAPHS * HIDDEN, B), B, 0, stream>>>(pooled, cnts, N_GRAPHS, HIDDEN);

  // ---- head: z = relu(pooled @ Wl1 + bl1); out = z @ Wl2 + bl2 ----
  {
    dim3 g(1, N_GRAPHS / 16);   // N = 256 exactly one block-strip
    k_gemm_wmma<<<g, 128, 0, stream>>>(pooled, Wtl1, bl1, z, N_GRAPHS, HID2, HIDDEN, 1);
  }
  {
    dim3 g(1, N_GRAPHS / 16);   // N = 128: waves 2,3 clamp, stores guarded
    k_gemm_wmma<<<g, 128, 0, stream>>>(z, Wtl2, bl2, out, N_GRAPHS, OUTF, HID2, 0);
  }
}
